// ImageCaptioningModel_51153060495387
// MI455X (gfx1250) — matrix operand order repaired and checked by reading, not compiled, not run on hardware
//
#include <hip/hip_runtime.h>
#include <hip/hip_bf16.h>
#include <cstdint>

// ---------------- problem constants ----------------
#define Vv 32000
#define Ee 256
#define Hh 512
#define Ff 512
#define Aa 256
#define Ll 49
#define Bb 64
#define Tt 32
#define KCAT 1280          // E + F + H = 256 + 512 + 512
#define G4H  2048          // 4*H

typedef __attribute__((ext_vector_type(16))) __bf16 v16bf;
typedef __attribute__((ext_vector_type(8)))  __bf16 v8bf;
typedef __attribute__((ext_vector_type(8)))  float  v8f;

__device__ __forceinline__ float sigm(float x) { return 1.0f / (1.0f + __expf(-x)); }

// A-fragment (16x32 bf16): lanes 0-15 -> M=lane, K = {k0..k0+7, k0+16..k0+23}
//                          lanes 16-31 -> M=lane-16, K = {k0+8..k0+15, k0+24..k0+31}
__device__ __forceinline__ v16bf load_fragA(const __bf16* __restrict__ base,
                                            int row, int ld, int k0, int lane) {
    const int half = (lane >> 4) & 1;
    const __bf16* p = base + (size_t)row * ld + k0 + half * 8;
    v8bf lo = *(const v8bf*)(p);
    v8bf hi = *(const v8bf*)(p + 16);
    v16bf r;
#pragma unroll
    for (int i = 0; i < 8; ++i) { r[i] = lo[i]; r[i + 8] = hi[i]; }
    return r;
}

// B-fragment (32x16 bf16), B supplied pre-transposed (N-major, K contiguous):
// lanes 0-15 -> N=lane, K = {k0..k0+7, k0+16..k0+23} ; lanes 16-31 -> N=lane-16, K = {k0+8..k0+15, k0+24..k0+31}
__device__ __forceinline__ v16bf load_fragB(const __bf16* __restrict__ base,
                                            int row, int ld, int k0, int lane) {
    const int half = (lane >> 4) & 1;
    const __bf16* p = base + (size_t)row * ld + k0 + half * 8;
    v8bf lo = *(const v8bf*)(p);
    v8bf hi = *(const v8bf*)(p + 16);
    v16bf r;
#pragma unroll
    for (int i = 0; i < 8; ++i) { r[i] = lo[i]; r[i + 8] = hi[i]; }
    return r;
}

// ---------------- prep kernels ----------------

// WcatT[n,k] (2048 x 1280) bf16 = transpose of [W_ih ; W_hh]
__global__ void prep_wcat_kernel(const float* __restrict__ W_ih,
                                 const float* __restrict__ W_hh,
                                 __bf16* __restrict__ WcatT) {
    int idx = blockIdx.x * blockDim.x + threadIdx.x;       // [0, 2048*1280)
    int n = idx / KCAT, k = idx % KCAT;
    float v = (k < (Ee + Ff)) ? W_ih[(size_t)k * G4H + n]
                              : W_hh[(size_t)(k - (Ee + Ff)) * G4H + n];
    WcatT[idx] = (__bf16)v;
}

// WoutT[n,k] (32000 x 512) bf16 = transpose of W_out (512 x 32000)
__global__ void prep_wout_kernel(const float* __restrict__ W_out,
                                 __bf16* __restrict__ WoutT) {
    size_t idx = (size_t)blockIdx.x * blockDim.x + threadIdx.x;  // [0, 32000*512)
    size_t n = idx / Hh, k = idx % Hh;
    WoutT[idx] = (__bf16)W_out[k * Vv + n];
}

// embedding gather -> bf16 (B*T, E)
__global__ void embed_kernel(const int* __restrict__ caption,
                             const float* __restrict__ embedding,
                             __bf16* __restrict__ embs) {
    int idx = blockIdx.x * blockDim.x + threadIdx.x;       // [0, B*T*E)
    int r = idx / Ee, e = idx % Ee;
    int tok = caption[r];
    embs[idx] = (__bf16)embedding[(size_t)tok * Ee + e];
}

// mean over L locations
__global__ void mean_kernel(const float* __restrict__ feats, float* __restrict__ mean) {
    int idx = blockIdx.x * blockDim.x + threadIdx.x;       // [0, B*F)
    int b = idx / Ff, f = idx % Ff;
    float s = 0.0f;
    for (int l = 0; l < Ll; ++l) s += feats[((size_t)(b * Ll + l)) * Ff + f];
    mean[idx] = s * (1.0f / Ll);
}

// h0 / c0 from mean feats; also seed the h-part of the bf16 A matrix
__global__ void init_hc_kernel(const float* __restrict__ mean,
                               const float* __restrict__ W_init_h, const float* __restrict__ b_init_h,
                               const float* __restrict__ W_init_c, const float* __restrict__ b_init_c,
                               float* __restrict__ h, float* __restrict__ c,
                               __bf16* __restrict__ Xh) {
    int idx = blockIdx.x * blockDim.x + threadIdx.x;       // [0, B*H)
    int b = idx / Hh, j = idx % Hh;
    float ah = b_init_h[j], ac = b_init_c[j];
    const float* m = mean + (size_t)b * Ff;
    for (int k = 0; k < Ff; ++k) {
        float mv = m[k];
        ah = fmaf(mv, W_init_h[(size_t)k * Hh + j], ah);
        ac = fmaf(mv, W_init_c[(size_t)k * Hh + j], ac);
    }
    h[idx] = ah; c[idx] = ac;
    Xh[(size_t)b * KCAT + (Ee + Ff) + j] = (__bf16)ah;
}

// feat_proj = image_feats @ W_feat + b_feat   (B*L, A)
__global__ void featproj_kernel(const float* __restrict__ feats,
                                const float* __restrict__ W_feat,
                                const float* __restrict__ b_feat,
                                float* __restrict__ fproj) {
    int idx = blockIdx.x * blockDim.x + threadIdx.x;       // [0, B*L*A)
    int r = idx / Aa, a = idx % Aa;
    float acc = b_feat[a];
    const float* fr = feats + (size_t)r * Ff;
    for (int k = 0; k < Ff; ++k) acc = fmaf(fr[k], W_feat[(size_t)k * Aa + a], acc);
    fproj[idx] = acc;
}

// ---------------- per-timestep kernels ----------------

// hW = h @ W_hid + b_hid   (B, A)
__global__ void hidproj_kernel(const float* __restrict__ h,
                               const float* __restrict__ W_hid,
                               const float* __restrict__ b_hid,
                               float* __restrict__ hW) {
    int idx = blockIdx.x * blockDim.x + threadIdx.x;       // [0, B*A)
    int b = idx / Aa, a = idx % Aa;
    float acc = b_hid[a];
    const float* hr = h + (size_t)b * Hh;
    for (int k = 0; k < Hh; ++k) acc = fmaf(hr[k], W_hid[(size_t)k * Aa + a], acc);
    hW[idx] = acc;
}

// fused attention: energy->scores->softmax->context, assemble bf16 [emb|ctx] row
__global__ void attn_kernel(const float* __restrict__ fproj,
                            const float* __restrict__ hW,
                            const float* __restrict__ feats,
                            const __bf16* __restrict__ embs,
                            const float* __restrict__ W_score,
                            const float* __restrict__ b_score,
                            __bf16* __restrict__ Xh,
                            float* __restrict__ alphas_out,
                            int t) {
    __shared__ float s_hW[256];
    __shared__ float s_red[256];
    __shared__ float s_sc[Ll];
    __shared__ float s_tmp;
    const int b = blockIdx.x, tid = threadIdx.x;

    s_hW[tid] = hW[b * Aa + tid];
    __syncthreads();

    for (int l = 0; l < Ll; ++l) {
        float v = tanhf(fproj[((size_t)(b * Ll + l)) * Aa + tid] + s_hW[tid]) * W_score[tid];
        s_red[tid] = v;
        __syncthreads();
        for (int s = 128; s > 0; s >>= 1) {
            if (tid < s) s_red[tid] += s_red[tid + s];
            __syncthreads();
        }
        if (tid == 0) s_sc[l] = s_red[0] + b_score[0];
        __syncthreads();
    }
    if (tid == 0) {
        float m = s_sc[0];
        for (int l = 1; l < Ll; ++l) m = fmaxf(m, s_sc[l]);
        s_tmp = m;
    }
    __syncthreads();
    float mx = s_tmp;
    if (tid < Ll) s_red[tid] = __expf(s_sc[tid] - mx);
    __syncthreads();
    if (tid == 0) {
        float s = 0.0f;
        for (int l = 0; l < Ll; ++l) s += s_red[l];
        s_tmp = 1.0f / s;
    }
    __syncthreads();
    if (tid < Ll) {
        float a = s_red[tid] * s_tmp;
        s_sc[tid] = a;
        alphas_out[((size_t)(b * Tt + t)) * Ll + tid] = a;
    }
    __syncthreads();
    // context (F=512, 256 threads -> 2 each), write bf16 into A-matrix row
    for (int f = tid; f < Ff; f += 256) {
        float ctx = 0.0f;
        for (int l = 0; l < Ll; ++l)
            ctx = fmaf(s_sc[l], feats[((size_t)(b * Ll + l)) * Ff + f], ctx);
        Xh[(size_t)b * KCAT + Ee + f] = (__bf16)ctx;
    }
    // embedding part
    Xh[(size_t)b * KCAT + tid] = embs[((size_t)(b * Tt + t)) * Ee + tid];
}

// gates = [X|h] @ WcatT^T : M=64, N=2048, K=1280 — WMMA, 16x64 tile / wave,
// software-pipelined (ping-pong register buffers) so loads overlap matrix ops.
__global__ void __launch_bounds__(256)
gates_gemm_kernel(const __bf16* __restrict__ Amat,
                  const __bf16* __restrict__ Bmat,
                  float* __restrict__ C) {
    const int lane = threadIdx.x & 31;
    const int w = blockIdx.x * (blockDim.x >> 5) + (threadIdx.x >> 5); // [0,128)
    const int mbase = (w & 3) * 16;
    const int nbase = (w >> 2) * 64;
    const int arow = mbase + (lane & 15);
    const int brow = nbase + (lane & 15);
    v8f acc[4] = {};

    v16bf aP, bP[4], aQ, bQ[4];
    aP = load_fragA(Amat, arow, KCAT, 0, lane);
#pragma unroll
    for (int s = 0; s < 4; ++s) bP[s] = load_fragB(Bmat, brow + s * 16, KCAT, 0, lane);

    // KCAT/32 = 40 iterations (even) -> 2x unrolled ping-pong
    for (int k0 = 0; k0 < KCAT; k0 += 64) {
        // stage k0+32 while computing on k0
        if (k0 + 32 < KCAT) {
            __builtin_prefetch(Amat + (size_t)arow * KCAT + k0 + 64, 0, 1);
            aQ = load_fragA(Amat, arow, KCAT, k0 + 32, lane);
#pragma unroll
            for (int s = 0; s < 4; ++s)
                bQ[s] = load_fragB(Bmat, brow + s * 16, KCAT, k0 + 32, lane);
        }
#pragma unroll
        for (int s = 0; s < 4; ++s)
            acc[s] = __builtin_amdgcn_wmma_f32_16x16x32_bf16(
                false, aP, false, bP[s], (short)0, acc[s], false, false);

        // stage k0+64 while computing on k0+32
        if (k0 + 64 < KCAT) {
            aP = load_fragA(Amat, arow, KCAT, k0 + 64, lane);
#pragma unroll
            for (int s = 0; s < 4; ++s)
                bP[s] = load_fragB(Bmat, brow + s * 16, KCAT, k0 + 64, lane);
        }
        if (k0 + 32 < KCAT) {
#pragma unroll
            for (int s = 0; s < 4; ++s)
                acc[s] = __builtin_amdgcn_wmma_f32_16x16x32_bf16(
                    false, aQ, false, bQ[s], (short)0, acc[s], false, false);
        }
    }

    const int half = lane >> 4, col0 = lane & 15;
#pragma unroll
    for (int s = 0; s < 4; ++s)
#pragma unroll
        for (int i = 0; i < 8; ++i)
            C[(size_t)(mbase + half * 8 + i) * G4H + nbase + s * 16 + col0] = acc[s][i];
}

// elementwise LSTM cell; deposits h_t into Hall (bf16) and next-step A matrix
__global__ void lstm_kernel(const float* __restrict__ gates,
                            const float* __restrict__ b_ih,
                            const float* __restrict__ b_hh,
                            float* __restrict__ h, float* __restrict__ c,
                            __bf16* __restrict__ Xh, __bf16* __restrict__ Hall,
                            int t) {
    int idx = blockIdx.x * blockDim.x + threadIdx.x;       // [0, B*H)
    int b = idx / Hh, j = idx % Hh;
    const float* g = gates + (size_t)b * G4H;
    float ig = sigm(g[j]            + b_ih[j]            + b_hh[j]);
    float fg = sigm(g[Hh + j]       + b_ih[Hh + j]       + b_hh[Hh + j]);
    float gg = tanhf(g[2 * Hh + j]  + b_ih[2 * Hh + j]   + b_hh[2 * Hh + j]);
    float og = sigm(g[3 * Hh + j]   + b_ih[3 * Hh + j]   + b_hh[3 * Hh + j]);
    float cn = fg * c[idx] + ig * gg;
    float hn = og * tanhf(cn);
    c[idx] = cn; h[idx] = hn;
    Xh[(size_t)b * KCAT + (Ee + Ff) + j] = (__bf16)hn;
    Hall[((size_t)(b * Tt + t)) * Hh + j] = (__bf16)hn;
}

// predictions = Hall(2048x512) @ WoutT^T(512x32000) + b_out — WMMA, 16x64/wave,
// software-pipelined (ping-pong register buffers).
__global__ void __launch_bounds__(256)
logits_gemm_kernel(const __bf16* __restrict__ Amat,
                   const __bf16* __restrict__ Bmat,
                   const float* __restrict__ b_out,
                   float* __restrict__ C) {
    const int lane = threadIdx.x & 31;
    const int wid = threadIdx.x >> 5;
    const int mbase = blockIdx.x * 16;                       // [0, 2048)
    const int ntile = blockIdx.y * 8 + wid;                  // 64-wide tiles
    if (ntile >= Vv / 64) return;
    const int nbase = ntile * 64;
    const int arow = mbase + (lane & 15);
    const int brow = nbase + (lane & 15);
    v8f acc[4] = {};

    v16bf aP, bP[4], aQ, bQ[4];
    aP = load_fragA(Amat, arow, Hh, 0, lane);
#pragma unroll
    for (int s = 0; s < 4; ++s) bP[s] = load_fragB(Bmat, brow + s * 16, Hh, 0, lane);

    // Hh/32 = 16 iterations (even) -> 2x unrolled ping-pong
    for (int k0 = 0; k0 < Hh; k0 += 64) {
        if (k0 + 32 < Hh) {
            __builtin_prefetch(Bmat + (size_t)brow * Hh + k0 + 64, 0, 1);
            aQ = load_fragA(Amat, arow, Hh, k0 + 32, lane);
#pragma unroll
            for (int s = 0; s < 4; ++s)
                bQ[s] = load_fragB(Bmat, brow + s * 16, Hh, k0 + 32, lane);
        }
#pragma unroll
        for (int s = 0; s < 4; ++s)
            acc[s] = __builtin_amdgcn_wmma_f32_16x16x32_bf16(
                false, aP, false, bP[s], (short)0, acc[s], false, false);

        if (k0 + 64 < Hh) {
            aP = load_fragA(Amat, arow, Hh, k0 + 64, lane);
#pragma unroll
            for (int s = 0; s < 4; ++s)
                bP[s] = load_fragB(Bmat, brow + s * 16, Hh, k0 + 64, lane);
        }
        if (k0 + 32 < Hh) {
#pragma unroll
            for (int s = 0; s < 4; ++s)
                acc[s] = __builtin_amdgcn_wmma_f32_16x16x32_bf16(
                    false, aQ, false, bQ[s], (short)0, acc[s], false, false);
        }
    }

    const int half = lane >> 4, col0 = lane & 15;
#pragma unroll
    for (int s = 0; s < 4; ++s) {
        int col = nbase + s * 16 + col0;
        float bias = b_out[col];
#pragma unroll
        for (int i = 0; i < 8; ++i)
            C[(size_t)(mbase + half * 8 + i) * Vv + col] = acc[s][i] + bias;
    }
}

// ---------------- launch ----------------
extern "C" void kernel_launch(void* const* d_in, const int* in_sizes, int n_in,
                              void* d_out, int out_size, void* d_ws, size_t ws_size,
                              hipStream_t stream) {
    const float* image_feats = (const float*)d_in[0];
    const int*   caption     = (const int*)d_in[1];
    const float* embedding   = (const float*)d_in[2];
    const float* W_init_h    = (const float*)d_in[3];
    const float* b_init_h    = (const float*)d_in[4];
    const float* W_init_c    = (const float*)d_in[5];
    const float* b_init_c    = (const float*)d_in[6];
    const float* W_feat      = (const float*)d_in[7];
    const float* b_feat      = (const float*)d_in[8];
    const float* W_hid       = (const float*)d_in[9];
    const float* b_hid       = (const float*)d_in[10];
    const float* W_score     = (const float*)d_in[11];
    const float* b_score     = (const float*)d_in[12];
    const float* W_ih        = (const float*)d_in[13];
    const float* b_ih        = (const float*)d_in[14];
    const float* W_hh        = (const float*)d_in[15];
    const float* b_hh        = (const float*)d_in[16];
    const float* W_out       = (const float*)d_in[17];
    const float* b_out       = (const float*)d_in[18];

    float* preds  = (float*)d_out;                               // (B,T,V)
    float* alphas = preds + (size_t)Bb * Tt * Vv;                // (B,T,L)

    // workspace carve-up (256B aligned)
    char* base = (char*)d_ws;
    size_t off = 0;
    auto carve = [&](size_t bytes) -> char* {
        char* p = base + off;
        off += (bytes + 255) & ~(size_t)255;
        return p;
    };
    float*  mean   = (float*)carve((size_t)Bb * Ff * 4);
    float*  fproj  = (float*)carve((size_t)Bb * Ll * Aa * 4);
    float*  h      = (float*)carve((size_t)Bb * Hh * 4);
    float*  c      = (float*)carve((size_t)Bb * Hh * 4);
    float*  hW     = (float*)carve((size_t)Bb * Aa * 4);
    float*  gates  = (float*)carve((size_t)Bb * G4H * 4);
    __bf16* embs   = (__bf16*)carve((size_t)Bb * Tt * Ee * 2);
    __bf16* Xh     = (__bf16*)carve((size_t)Bb * KCAT * 2);
    __bf16* Hall   = (__bf16*)carve((size_t)Bb * Tt * Hh * 2);
    __bf16* WcatT  = (__bf16*)carve((size_t)G4H * KCAT * 2);
    __bf16* WoutT  = (__bf16*)carve((size_t)Vv * Hh * 2);

    // one-time prep (re-recorded every call: deterministic)
    prep_wcat_kernel<<<(G4H * KCAT) / 256, 256, 0, stream>>>(W_ih, W_hh, WcatT);
    prep_wout_kernel<<<((size_t)Vv * Hh) / 256, 256, 0, stream>>>(W_out, WoutT);
    embed_kernel<<<(Bb * Tt * Ee) / 256, 256, 0, stream>>>(caption, embedding, embs);
    mean_kernel<<<(Bb * Ff) / 256, 256, 0, stream>>>(image_feats, mean);
    init_hc_kernel<<<(Bb * Hh) / 256, 256, 0, stream>>>(mean, W_init_h, b_init_h,
                                                        W_init_c, b_init_c, h, c, Xh);
    featproj_kernel<<<(Bb * Ll * Aa) / 256, 256, 0, stream>>>(image_feats, W_feat, b_feat, fproj);

    // sequential recurrence
    for (int t = 0; t < Tt; ++t) {
        hidproj_kernel<<<(Bb * Aa) / 256, 256, 0, stream>>>(h, W_hid, b_hid, hW);
        attn_kernel<<<Bb, 256, 0, stream>>>(fproj, hW, image_feats, embs,
                                            W_score, b_score, Xh, alphas, t);
        gates_gemm_kernel<<<16, 256, 0, stream>>>(Xh, WcatT, gates);
        lstm_kernel<<<(Bb * Hh) / 256, 256, 0, stream>>>(gates, b_ih, b_hh, h, c, Xh, Hall, t);
    }

    // one big logits GEMM: (B*T, H) @ (H, V) + b_out
    dim3 lgrid((Bb * Tt) / 16, (Vv / 64 + 7) / 8);
    logits_gemm_kernel<<<lgrid, 256, 0, stream>>>(Hall, WoutT, b_out, preds);
}
